// VisionMamba_72387378807131
// MI455X (gfx1250) — matrix operand order repaired and checked by reading, not compiled, not run on hardware
//
#include <hip/hip_runtime.h>
#include <hip/hip_bf16.h>
#include <math.h>

// ---------------------------------------------------------------------------
// Vision Mamba forward for MI455X (gfx1250, wave32, WMMA).
// All dense einsums -> V_WMMA_F32_16X16X32_F16 (f16 operands, f32 accumulate).
// Branch-free K-loop: out-of-tile rows are clamped (stores are guarded), and
// the K=24 dt-projection is zero-padded to K=32 so no K guards are needed.
// ---------------------------------------------------------------------------

typedef __attribute__((ext_vector_type(16))) _Float16 v16h;
typedef __attribute__((ext_vector_type(8)))  float    v8f;

#define D_MODEL 384
#define D_INNER 768
#define N_STATE 16
#define R_RANK  24
#define R_PAD   32                     // dt weights zero-padded 24 -> 32
#define DEPTH_L 12
#define K_CONV  4
#define PATCH   16
#define BATCH   2
#define IMG_SZ  224
#define GRID_P  14                     // IMG/PATCH
#define NPATCH  196                    // 14*14
#define SEQ_L   197                    // NPATCH + cls
#define BL_TOT  (BATCH * SEQ_L)        // 394
#define PROJ_W  (R_RANK + 2 * N_STATE) // 56
#define K_PATCH (3 * PATCH * PATCH)    // 768
#define EPS_RMS 1e-5f

__device__ __forceinline__ float sigmoidf_(float x) { return 1.0f / (1.0f + __expf(-x)); }
__device__ __forceinline__ float softplusf_(float x) { return (x > 20.0f) ? x : log1pf(__expf(x)); }

// ---------------------------------------------------------------------------
// WMMA GEMM:  C[M,N] = A[M,K] (row-major f32) * W[N,K]^T (row-major f32)
// K is compile-time and a multiple of 32. act: 0 none, 1 +bias, 2 +bias,softplus
// One wave -> one 16x16 tile, 4 waves per block along N. Inner loop has no
// branches: OOB rows/cols are clamped to row 0 (garbage discarded at store).
// ---------------------------------------------------------------------------
template<int K>
__global__ __launch_bounds__(128) void wmma_gemm_kernel(
    const float* __restrict__ A, int lda,
    const float* __restrict__ W, int ldw,
    float* __restrict__ C, int ldc,
    const float* __restrict__ bias,
    int M, int N, int act)
{
    const int lane  = threadIdx.x & 31;
    const int wid   = threadIdx.x >> 5;
    const int mtile = blockIdx.y << 4;
    const int ntile = (blockIdx.x * 4 + wid) << 4;
    if (ntile >= N) return;                      // wave-uniform exit

    const int kh  = lane >> 4;                   // half-wave select
    const int l16 = lane & 15;

    const int arow = mtile + l16;                // A row striped per lane
    const int bcol = ntile + l16;                // W row == output column
    const float* Ar = A + (size_t)(arow < M ? arow : 0) * lda;
    const float* Wr = W + (size_t)(bcol < N ? bcol : 0) * ldw;

    v8f acc = {};
#pragma unroll 2
    for (int k0 = 0; k0 < K; k0 += 32) {
        // A fragment (ISA 7.12.2): lane kh=0 carries K = k0+0..7 and k0+16..23,
        // kh=1 carries K = k0+8..15 and k0+24..31 -> two contiguous 8-runs.
        float va[16], vb[16];
        *(float4*)(va + 0)  = *(const float4*)(Ar + k0 + kh * 8);
        *(float4*)(va + 4)  = *(const float4*)(Ar + k0 + kh * 8 + 4);
        *(float4*)(va + 8)  = *(const float4*)(Ar + k0 + 16 + kh * 8);
        *(float4*)(va + 12) = *(const float4*)(Ar + k0 + 16 + kh * 8 + 4);
        // B fragment: lane holds column n, same K runs as A (k0+8kh.., k0+16+8kh..).
        *(float4*)(vb + 0)  = *(const float4*)(Wr + k0 + kh * 8);
        *(float4*)(vb + 4)  = *(const float4*)(Wr + k0 + kh * 8 + 4);
        *(float4*)(vb + 8)  = *(const float4*)(Wr + k0 + 16 + kh * 8);
        *(float4*)(vb + 12) = *(const float4*)(Wr + k0 + 16 + kh * 8 + 4);

        __builtin_prefetch(Ar + k0 + 32, 0, 1);   // global_prefetch_b8
        __builtin_prefetch(Wr + k0 + 32, 0, 1);

        v16h af, bf;
#pragma unroll
        for (int j = 0; j < 16; ++j) {
            af[j] = (_Float16)va[j];
            bf[j] = (_Float16)vb[j];
        }
        acc = __builtin_amdgcn_wmma_f32_16x16x32_f16(
                  false, af, false, bf, (short)0, acc, false, false);
    }

    const int n = ntile + l16;
    if (n < N) {
        const float bv = bias ? bias[n] : 0.0f;
#pragma unroll
        for (int r = 0; r < 8; ++r) {
            const int m = mtile + kh * 8 + r;     // C layout: VGPR r, half kh
            if (m < M) {
                float v = acc[r];
                if (act >= 1) v += bv;
                if (act == 2) v = softplusf_(v);
                C[(size_t)m * ldc + n] = v;
            }
        }
    }
}

// zero-pad dt weights [DEPTH][DI][24] -> [DEPTH][DI][32]
__global__ void pad_dtw_kernel(const float* __restrict__ src, float* __restrict__ dst)
{
    int idx = blockIdx.x * blockDim.x + threadIdx.x;
    const int total = DEPTH_L * D_INNER * R_PAD;
    if (idx >= total) return;
    int k = idx % R_PAD;
    int row = idx / R_PAD;
    dst[idx] = (k < R_RANK) ? src[(size_t)row * R_RANK + k] : 0.0f;
}

// ---------------------------------------------------------------------------
// im2col for 16x16 non-overlapping patches: Acol[b*196+p, c*256+py*16+px]
// ---------------------------------------------------------------------------
__global__ void im2col_kernel(const float* __restrict__ x, float* __restrict__ Acol)
{
    int idx = blockIdx.x * blockDim.x + threadIdx.x;
    const int total = BATCH * NPATCH * K_PATCH;
    if (idx >= total) return;
    int k = idx % K_PATCH;
    int t = idx / K_PATCH;
    int b = t / NPATCH;
    int p = t % NPATCH;
    int ph = p / GRID_P, pw = p % GRID_P;
    int c  = k / (PATCH * PATCH);
    int r  = k % (PATCH * PATCH);
    int py = r / PATCH, px = r % PATCH;
    Acol[idx] = x[(((size_t)b * 3 + c) * IMG_SZ + (ph * PATCH + py)) * IMG_SZ
                  + (pw * PATCH + px)];
}

// h[b,l,d] = (l==0 ? cls[d] : tokens[b*196+l-1, d]) + pos[l,d]
__global__ void assemble_h_kernel(const float* __restrict__ tokens,
                                  const float* __restrict__ cls,
                                  const float* __restrict__ pos,
                                  float* __restrict__ h)
{
    int idx = blockIdx.x * blockDim.x + threadIdx.x;
    const int total = BL_TOT * D_MODEL;
    if (idx >= total) return;
    int d = idx % D_MODEL;
    int row = idx / D_MODEL;
    int b = row / SEQ_L, l = row % SEQ_L;
    float v = (l == 0) ? cls[d]
                       : tokens[((size_t)b * NPATCH + (l - 1)) * D_MODEL + d];
    h[idx] = v + pos[(size_t)l * D_MODEL + d];
}

// res = hin (+res); hout = res * rsqrt(mean(res^2)+eps) * w. One block per row.
__global__ __launch_bounds__(128) void add_rmsnorm_kernel(
    const float* __restrict__ hin, float* __restrict__ res,
    float* __restrict__ hout, const float* __restrict__ w, int add_res)
{
    const int row = blockIdx.x;
    const float* hp = hin + (size_t)row * D_MODEL;
    float* rp = res + (size_t)row * D_MODEL;
    __shared__ float sred[128];
    float vals[3];
    float ss = 0.0f;
#pragma unroll
    for (int j = 0; j < 3; ++j) {
        int d = threadIdx.x + j * 128;
        float v = hp[d];
        if (add_res) v += rp[d];
        vals[j] = v;
        ss += v * v;
    }
    sred[threadIdx.x] = ss;
    __syncthreads();
    for (int s = 64; s > 0; s >>= 1) {
        if (threadIdx.x < s) sred[threadIdx.x] += sred[threadIdx.x + s];
        __syncthreads();
    }
    const float scale = rsqrtf(sred[0] / (float)D_MODEL + EPS_RMS);
#pragma unroll
    for (int j = 0; j < 3; ++j) {
        int d = threadIdx.x + j * 128;
        rp[d] = vals[j];
        hout[(size_t)row * D_MODEL + d] = vals[j] * scale * w[d];
    }
}

// xcrev[b,l,i] = xz[b, L-1-l, i]   (xc = first DI columns of xz)
__global__ void reverse_seq_kernel(const float* __restrict__ xz,
                                   float* __restrict__ xcrev)
{
    int idx = blockIdx.x * blockDim.x + threadIdx.x;
    const int total = BL_TOT * D_INNER;
    if (idx >= total) return;
    int i = idx % D_INNER;
    int row = idx / D_INNER;
    int b = row / SEQ_L, l = row % SEQ_L;
    xcrev[idx] = xz[((size_t)b * SEQ_L + (SEQ_L - 1 - l)) * (2 * D_INNER) + i];
}

// depthwise causal conv (K=4, left pad 3) + SiLU, per (b,l,i)
__global__ void conv_silu_kernel(const float* __restrict__ X, int xstride,
                                 const float* __restrict__ cw,
                                 const float* __restrict__ cb,
                                 float* __restrict__ U)
{
    int idx = blockIdx.x * blockDim.x + threadIdx.x;
    const int total = BL_TOT * D_INNER;
    if (idx >= total) return;
    int i = idx % D_INNER;
    int row = idx / D_INNER;
    int l = row % SEQ_L;
    int base = row - l;
    float s = cb[i];
#pragma unroll
    for (int k = 0; k < K_CONV; ++k) {
        int ll = l - (K_CONV - 1) + k;
        if (ll >= 0) s += cw[i * K_CONV + k] * X[(size_t)(base + ll) * xstride + i];
    }
    U[(size_t)row * D_INNER + i] = s * sigmoidf_(s);
}

// selective scan: one thread per (b, channel); 16-float state in registers
__global__ __launch_bounds__(256) void selective_scan_kernel(
    const float* __restrict__ U, const float* __restrict__ DT,
    const float* __restrict__ PROJ, const float* __restrict__ Alog,
    const float* __restrict__ Dsk, float* __restrict__ Y)
{
    int t = blockIdx.x * blockDim.x + threadIdx.x;
    if (t >= BATCH * D_INNER) return;
    int b = t / D_INNER, i = t % D_INNER;
    float Av[N_STATE];
#pragma unroll
    for (int n = 0; n < N_STATE; ++n)
        Av[n] = -__expf(Alog[(size_t)i * N_STATE + n]);
    const float Dp = Dsk[i];
    float st[N_STATE];
#pragma unroll
    for (int n = 0; n < N_STATE; ++n) st[n] = 0.0f;

    for (int l = 0; l < SEQ_L; ++l) {
        const size_t row = (size_t)b * SEQ_L + l;
        const float dt = DT[row * D_INNER + i];
        const float u  = U[row * D_INNER + i];
        const float du = dt * u;
        const float* pr = PROJ + row * PROJ_W;
        float y = u * Dp;
#pragma unroll
        for (int n = 0; n < N_STATE; ++n) {
            st[n] = st[n] * __expf(dt * Av[n]) + du * pr[R_RANK + n];
            y += st[n] * pr[R_RANK + N_STATE + n];
        }
        Y[row * D_INNER + i] = y;
    }
}

// y = (y_f + reverse(y_brev)) * silu(z);  z = xz[:, DI:]
__global__ void gate_combine_kernel(const float* __restrict__ xz,
                                    const float* __restrict__ yf,
                                    const float* __restrict__ ybrev,
                                    float* __restrict__ yc)
{
    int idx = blockIdx.x * blockDim.x + threadIdx.x;
    const int total = BL_TOT * D_INNER;
    if (idx >= total) return;
    int i = idx % D_INNER;
    int row = idx / D_INNER;
    int b = row / SEQ_L, l = row % SEQ_L;
    float z  = xz[(size_t)row * (2 * D_INNER) + D_INNER + i];
    float yb = ybrev[((size_t)b * SEQ_L + (SEQ_L - 1 - l)) * D_INNER + i];
    yc[idx] = (yf[idx] + yb) * (z * sigmoidf_(z));
}

// ---------------------------------------------------------------------------
template<int K>
static inline void launch_gemm(const float* A, int lda, const float* Wt, int ldw,
                               float* C, int ldc, const float* bias,
                               int M, int N, int act, hipStream_t stream)
{
    dim3 grid((unsigned)(((N + 15) / 16 + 3) / 4), (unsigned)((M + 15) / 16));
    hipLaunchKernelGGL((wmma_gemm_kernel<K>), grid, dim3(128), 0, stream,
                       A, lda, Wt, ldw, C, ldc, bias, M, N, act);
}

extern "C" void kernel_launch(void* const* d_in, const int* in_sizes, int n_in,
                              void* d_out, int out_size, void* d_ws, size_t ws_size,
                              hipStream_t stream)
{
    (void)in_sizes; (void)n_in; (void)out_size; (void)ws_size;
    const float* x        = (const float*)d_in[0];
    const float* patch_w  = (const float*)d_in[1];
    const float* patch_b  = (const float*)d_in[2];
    const float* cls_tok  = (const float*)d_in[3];
    const float* pos_emb  = (const float*)d_in[4];
    const float* norm_w   = (const float*)d_in[5];
    const float* in_proj  = (const float*)d_in[6];
    const float* conv_w   = (const float*)d_in[7];
    const float* conv_b   = (const float*)d_in[8];
    const float* x_proj   = (const float*)d_in[9];
    const float* dtw      = (const float*)d_in[10];
    const float* dtb      = (const float*)d_in[11];
    const float* A_log    = (const float*)d_in[12];
    const float* D_skip   = (const float*)d_in[13];
    const float* conv_w_b = (const float*)d_in[14];
    const float* conv_b_b = (const float*)d_in[15];
    const float* x_proj_b = (const float*)d_in[16];
    const float* dtw_b    = (const float*)d_in[17];
    const float* dtb_b    = (const float*)d_in[18];
    const float* A_log_b  = (const float*)d_in[19];
    const float* D_skip_b = (const float*)d_in[20];
    const float* out_proj = (const float*)d_in[21];
    const float* norm_f_w = (const float*)d_in[22];

    float* w = (float*)d_ws;
    float* hbuf  = w; w += (size_t)BL_TOT * D_MODEL;
    float* res   = w; w += (size_t)BL_TOT * D_MODEL;
    float* hn    = w; w += (size_t)BL_TOT * D_MODEL;
    float* xz    = w; w += (size_t)BL_TOT * 2 * D_INNER;
    float* xcrev = w; w += (size_t)BL_TOT * D_INNER;
    float* u_f   = w; w += (size_t)BL_TOT * D_INNER;
    float* u_b   = w; w += (size_t)BL_TOT * D_INNER;
    float* pj_f  = w; w += (size_t)BL_TOT * PROJ_W;
    float* pj_b  = w; w += (size_t)BL_TOT * PROJ_W;
    float* dt_fb = w; w += (size_t)BL_TOT * D_INNER;
    float* dt_bb = w; w += (size_t)BL_TOT * D_INNER;
    float* y_f   = w; w += (size_t)BL_TOT * D_INNER;
    float* y_br  = w; w += (size_t)BL_TOT * D_INNER;
    float* ycmb  = w; w += (size_t)BL_TOT * D_INNER;
    float* Acol  = w; w += (size_t)BATCH * NPATCH * K_PATCH;
    float* toks  = w; w += (size_t)BATCH * NPATCH * D_MODEL;
    float* dtwp  = w; w += (size_t)DEPTH_L * D_INNER * R_PAD;  // padded dt w (fwd)
    float* dtwpb = w; w += (size_t)DEPTH_L * D_INNER * R_PAD;  // padded dt w (bwd)

    auto blocks = [](int total) { return dim3((unsigned)((total + 255) / 256)); };

    // ---- zero-pad dt weights so the dt GEMM needs no K guards ------------
    {
        const int tot = DEPTH_L * D_INNER * R_PAD;
        hipLaunchKernelGGL(pad_dtw_kernel, blocks(tot), dim3(256), 0, stream, dtw,   dtwp);
        hipLaunchKernelGGL(pad_dtw_kernel, blocks(tot), dim3(256), 0, stream, dtw_b, dtwpb);
    }

    // ---- patch embedding --------------------------------------------------
    hipLaunchKernelGGL(im2col_kernel, blocks(BATCH * NPATCH * K_PATCH), dim3(256),
                       0, stream, x, Acol);
    launch_gemm<K_PATCH>(Acol, K_PATCH, patch_w, K_PATCH, toks, D_MODEL, patch_b,
                         BATCH * NPATCH, D_MODEL, 1, stream);
    hipLaunchKernelGGL(assemble_h_kernel, blocks(BL_TOT * D_MODEL), dim3(256),
                       0, stream, toks, cls_tok, pos_emb, hbuf);

    const int tDI = BL_TOT * D_INNER;
    // ---- mamba layers -----------------------------------------------------
    for (int layer = 0; layer < DEPTH_L; ++layer) {
        hipLaunchKernelGGL(add_rmsnorm_kernel, dim3(BL_TOT), dim3(128), 0, stream,
                           hbuf, res, hn, norm_w + (size_t)layer * D_MODEL,
                           layer == 0 ? 0 : 1);

        launch_gemm<D_MODEL>(hn, D_MODEL,
                             in_proj + (size_t)layer * 2 * D_INNER * D_MODEL, D_MODEL,
                             xz, 2 * D_INNER, nullptr, BL_TOT, 2 * D_INNER, 0, stream);

        hipLaunchKernelGGL(reverse_seq_kernel, blocks(tDI), dim3(256), 0, stream,
                           xz, xcrev);

        hipLaunchKernelGGL(conv_silu_kernel, blocks(tDI), dim3(256), 0, stream,
                           xz, 2 * D_INNER,
                           conv_w + (size_t)layer * D_INNER * K_CONV,
                           conv_b + (size_t)layer * D_INNER, u_f);
        hipLaunchKernelGGL(conv_silu_kernel, blocks(tDI), dim3(256), 0, stream,
                           xcrev, D_INNER,
                           conv_w_b + (size_t)layer * D_INNER * K_CONV,
                           conv_b_b + (size_t)layer * D_INNER, u_b);

        launch_gemm<D_INNER>(u_f, D_INNER,
                             x_proj + (size_t)layer * PROJ_W * D_INNER, D_INNER,
                             pj_f, PROJ_W, nullptr, BL_TOT, PROJ_W, 0, stream);
        launch_gemm<D_INNER>(u_b, D_INNER,
                             x_proj_b + (size_t)layer * PROJ_W * D_INNER, D_INNER,
                             pj_b, PROJ_W, nullptr, BL_TOT, PROJ_W, 0, stream);

        launch_gemm<R_PAD>(pj_f, PROJ_W, dtwp + (size_t)layer * D_INNER * R_PAD, R_PAD,
                           dt_fb, D_INNER, dtb + (size_t)layer * D_INNER,
                           BL_TOT, D_INNER, 2, stream);
        launch_gemm<R_PAD>(pj_b, PROJ_W, dtwpb + (size_t)layer * D_INNER * R_PAD, R_PAD,
                           dt_bb, D_INNER, dtb_b + (size_t)layer * D_INNER,
                           BL_TOT, D_INNER, 2, stream);

        hipLaunchKernelGGL(selective_scan_kernel,
                           dim3((BATCH * D_INNER + 255) / 256), dim3(256), 0, stream,
                           u_f, dt_fb, pj_f,
                           A_log + (size_t)layer * D_INNER * N_STATE,
                           D_skip + (size_t)layer * D_INNER, y_f);
        hipLaunchKernelGGL(selective_scan_kernel,
                           dim3((BATCH * D_INNER + 255) / 256), dim3(256), 0, stream,
                           u_b, dt_bb, pj_b,
                           A_log_b + (size_t)layer * D_INNER * N_STATE,
                           D_skip_b + (size_t)layer * D_INNER, y_br);

        hipLaunchKernelGGL(gate_combine_kernel, blocks(tDI), dim3(256), 0, stream,
                           xz, y_f, y_br, ycmb);

        launch_gemm<D_INNER>(ycmb, D_INNER,
                             out_proj + (size_t)layer * D_MODEL * D_INNER, D_INNER,
                             hbuf, D_MODEL, nullptr, BL_TOT, D_MODEL, 0, stream);
    }

    // ---- final norm -> d_out ---------------------------------------------
    hipLaunchKernelGGL(add_rmsnorm_kernel, dim3(BL_TOT), dim3(128), 0, stream,
                       hbuf, res, (float*)d_out, norm_f_w, 1);
}
